// DeformConv2d_50371376447821
// MI455X (gfx1250) — matrix operand order repaired and checked by reading, not compiled, not run on hardware
//
#include <hip/hip_runtime.h>
#include <hip/hip_bf16.h>

// ---------------------------------------------------------------------------
// Deformable conv (feat_group=False) on MI455X / gfx1250.
//   x:   [4, 4, 256, 36, 36] f32   (m, g, c, y, x), b = m*4+g
//   flt: [1024, 256, 3, 3]  f32    (oc = g*256+o, c, i, j)
//   off: [16, 2]            f32
//   out: [4, 1024, 36, 36]  f32
// GEMM per (m,g): M=256 (o), K=2304 (tap*256+c), N=1296 (y*36+x)
// B (bilinear im2col) generated on the fly into double-buffered LDS in bf16.
// Bilinear weights/offsets cached in VGPRs per tap (refreshed every 8 K-steps).
// ---------------------------------------------------------------------------

typedef __attribute__((ext_vector_type(16))) __bf16  v16bf;
typedef __attribute__((ext_vector_type(8)))  float   v8f;
typedef __attribute__((ext_vector_type(4)))  unsigned int u32x4;

union Frag { u32x4 q[2]; v16bf v; };

__device__ __forceinline__ unsigned short f2bf(float f) {
    unsigned int u = __float_as_uint(f);
    unsigned int r = u + 0x7FFFu + ((u >> 16) & 1u);   // round-to-nearest-even
    return (unsigned short)(r >> 16);
}

__device__ __forceinline__ unsigned int pack_bf16(float lo, float hi) {
#if __has_builtin(__builtin_amdgcn_cvt_pk_bf16_f32)
    auto r = __builtin_amdgcn_cvt_pk_bf16_f32(lo, hi);  // v_cvt_pk_bf16_f32
    union { decltype(r) v; unsigned int u; } cv; cv.v = r;
    return cv.u;
#else
    return (unsigned int)f2bf(lo) | ((unsigned int)f2bf(hi) << 16);
#endif
}

__device__ __forceinline__ void sched_fence() {
#if __has_builtin(__builtin_amdgcn_sched_barrier)
    __builtin_amdgcn_sched_barrier(0);                 // no reordering across
#endif
}

#define H 36
#define C_IN 256
#define TAPS 9
#define KTOT 2304          // 256 * 9
#define NPOS 1296          // 36*36
#define NT   48            // N macro-tile (positions per block)
#define BSTRIDE 40         // bf16 elems per LDS B column (32 + 8 pad = 80B, 16B aligned)

// --- Filter fp32 -> bf16, relayout to A[g*256+o][tap*256+c] ---------------
__global__ __launch_bounds__(256) void filter_to_bf16(const float* __restrict__ tf,
                                                      unsigned short* __restrict__ A) {
    int idx = blockIdx.x * 256 + threadIdx.x;          // 1024*2304 total
    if (idx >= 1024 * KTOT) return;
    int oc  = idx / KTOT;
    int k   = idx - oc * KTOT;
    int tap = k >> 8;                                  // k / 256
    int c   = k & 255;
    A[idx] = f2bf(tf[(oc * C_IN + c) * TAPS + tap]);
}

// --- Fused deform-sample + WMMA GEMM --------------------------------------
// grid: (27 N-tiles, 2 M-tiles, 16 b) ; block: 256 threads (8 wave32)
__global__ __launch_bounds__(256, 4) void deform_wmma(const float* __restrict__ xg,
                                                      const unsigned short* __restrict__ Abf,
                                                      const float* __restrict__ off,
                                                      float* __restrict__ outp) {
    __shared__ unsigned short Bl[2][NT * BSTRIDE];     // 2 x 3.75 KB, double-buffered
    // Packed bilinear tables: {w_left_bits, w_right_bits, byteoff_L, byteoff_R}
    __shared__ u32x4 tX[108], tY[108];

    const int tid  = threadIdx.x;
    const int p0   = blockIdx.x * NT;                  // first position of N-tile
    const int oT   = blockIdx.y;                       // M tile (0..1)
    const int b    = blockIdx.z;                       // m*4+g
    const int g    = b & 3;
    const int m    = b >> 2;

    const float sx = 3.0f / off[b * 2 + 0];
    const float sy = 3.0f / off[b * 2 + 1];

    // Per-axis bilinear tables: index (pos*3 + tap_axis); zero padding folded
    // into weight==0; indices stored premultiplied to BYTE offsets.
    if (tid < 216) {
        const int axis = (tid >= 108);
        const int e    = axis ? tid - 108 : tid;
        const int pos  = e / 3;
        const int i    = e - pos * 3;
        const float s  = axis ? sy : sx;
        const float p  = (float)(pos + 1) + (float)(i - 1) * s;
        const float pc = fminf(fmaxf(p, 0.0f), 37.0f);
        const float fl = floorf(p);
        const float ql = fminf(fmaxf(fl,        0.0f), 37.0f);
        const float qr = fminf(fmaxf(fl + 1.0f, 0.0f), 37.0f);
        const float gl = 1.0f + (ql - pc);
        const float gr = 1.0f - (qr - pc);
        const int il = (int)ql - 1, ir = (int)qr - 1;
        const bool vl = (il >= 0) & (il < H);
        const bool vr = (ir >= 0) & (ir < H);
        const int stride = axis ? 4 : H * 4;           // col vs row byte stride
        u32x4 q;
        q[0] = vl ? __float_as_uint(gl) : 0u;          // 0u == 0.0f bits
        q[1] = vr ? __float_as_uint(gr) : 0u;
        q[2] = vl ? (unsigned)(il * stride) : 0u;
        q[3] = vr ? (unsigned)(ir * stride) : 0u;
        if (axis) tY[e] = q; else tX[e] = q;
    }
    __syncthreads();

    const int lane = tid & 31;
    const int wv   = tid >> 5;                         // wave id 0..7 -> M strip
    const int l16  = lane & 15;
    const int half = lane >> 4;

    const char* xb = (const char*)(xg + (size_t)b * C_IN * NPOS);

    // B-tile fill: 3 passes/thread, 2 adjacent channels per pass, 16 thr/column.
    const int fn_off = tid >> 4;                       // column within 16-group
    const int fkc    = (tid & 15) * 2;                 // even channel pair
    const int fkcOff = fkc * (NPOS * 4);               // byte offset of channel plane

    // Per-tap register cache: 4 bilinear weights + 4 corner byte-offsets per pass.
    float cw[3][4];
    int   co[3][4];
    auto loadTables = [&](int tapf) {
        const int ti = tapf / 3;
        const int tj = tapf - ti * 3;
        #pragma unroll
        for (int p = 0; p < 3; ++p) {
            const int pos = p0 + p * 16 + fn_off;
            const int py  = pos / H;
            const int pxc = pos - py * H;
            const u32x4 qx = tX[py * 3 + ti];
            const u32x4 qy = tY[pxc * 3 + tj];
            const float wxl = __uint_as_float(qx[0]), wxr = __uint_as_float(qx[1]);
            const float wyl = __uint_as_float(qy[0]), wyr = __uint_as_float(qy[1]);
            cw[p][0] = wxl * wyl; cw[p][1] = wxl * wyr;
            cw[p][2] = wxr * wyl; cw[p][3] = wxr * wyr;
            co[p][0] = fkcOff + (int)qx[2] + (int)qy[2];
            co[p][1] = fkcOff + (int)qx[2] + (int)qy[3];
            co[p][2] = fkcOff + (int)qx[3] + (int)qy[2];
            co[p][3] = fkcOff + (int)qx[3] + (int)qy[3];
        }
    };

    // Steady-state fill: 24 saddr gathers + packed FMAs + 3 ds_store_b32.
    auto fill = [&](int kkf, unsigned short* Bbuf) {
        const char* xc0 = xb + (kkf & 7) * (32 * NPOS * 4);  // scalar plane base
        #pragma unroll
        for (int p = 0; p < 3; ++p) {
            const int n = p * 16 + fn_off;
#define LDX(o) (*(const float*)(xc0 + (o)))
            const float v0 = cw[p][0] * LDX(co[p][0]) + cw[p][1] * LDX(co[p][1])
                           + cw[p][2] * LDX(co[p][2]) + cw[p][3] * LDX(co[p][3]);
            const float v1 = cw[p][0] * LDX(co[p][0] + NPOS * 4)
                           + cw[p][1] * LDX(co[p][1] + NPOS * 4)
                           + cw[p][2] * LDX(co[p][2] + NPOS * 4)
                           + cw[p][3] * LDX(co[p][3] + NPOS * 4);
#undef LDX
            *(unsigned int*)&Bbuf[n * BSTRIDE + fkc] = pack_bf16(v0, v1);
        }
    };

    v8f acc[3] = {};                                   // 3 N-subtiles, 16x16 f32 each
    const unsigned short* Arow =
        Abf + (size_t)(g * 256 + oT * 128 + wv * 16 + l16) * KTOT;

    loadTables(0);
    fill(0, Bl[0]);

    // Software-pipelined A fragment (16-bit A layout: half K pairs 0..7/16..23)
    Frag fa;
    fa.q[0] = *(const u32x4*)(Arow + half * 8);
    fa.q[1] = *(const u32x4*)(Arow + half * 8 + 16);
    __syncthreads();

    for (int kk = 0; kk < 72; ++kk) {
        const unsigned short* cur = Bl[kk & 1];

        // issue next A fragment early; its latency hides under the WMMAs
        Frag fan;
        if (kk + 1 < 72) {
            fan.q[0] = *(const u32x4*)(Arow + (kk + 1) * 32 + half * 8);
            fan.q[1] = *(const u32x4*)(Arow + (kk + 1) * 32 + half * 8 + 16);
        }

        // preload ALL B fragments; fence so the scheduler cannot sink these
        // ds_loads below the WMMAs (keeps dscnt waits partial, latency shared)
        Frag fb[3];
        #pragma unroll
        for (int t = 0; t < 3; ++t) {
            const unsigned short* colp = &cur[(t * 16 + l16) * BSTRIDE + half * 8];
            fb[t].q[0] = *(const u32x4*)(colp);
            fb[t].q[1] = *(const u32x4*)(colp + 16);
        }
        sched_fence();
        #pragma unroll
        for (int t = 0; t < 3; ++t)
            acc[t] = __builtin_amdgcn_wmma_f32_16x16x32_bf16(
                false, fa.v, false, fb[t].v, (short)0, acc[t], false, false);

        // overlap: generate next K-step's B tile into the other buffer
        if (kk + 1 < 72) {
            if (((kk + 1) & 7) == 0) loadTables((kk + 1) >> 3);  // uniform branch
            fill(kk + 1, Bl[(kk + 1) & 1]);
            fa = fan;
        }
        __syncthreads();
    }

    // ---- store: D layout lane=N col, VGPR v -> M row v + 8*half
    const size_t obase = ((size_t)(m * 1024 + g * 256)) * NPOS + (size_t)p0;
    #pragma unroll
    for (int t = 0; t < 3; ++t) {
        const int n = t * 16 + l16;
        #pragma unroll
        for (int v = 0; v < 8; ++v) {
            const int o = oT * 128 + wv * 16 + v + 8 * half;
            outp[obase + (size_t)o * NPOS + n] = acc[t][v];
        }
    }
}

extern "C" void kernel_launch(void* const* d_in, const int* in_sizes, int n_in,
                              void* d_out, int out_size, void* d_ws, size_t ws_size,
                              hipStream_t stream) {
    const float* x   = (const float*)d_in[0];
    const float* tf  = (const float*)d_in[1];
    const float* off = (const float*)d_in[2];
    float* outp      = (float*)d_out;
    unsigned short* Abf = (unsigned short*)d_ws;       // 1024*2304*2 = 4.5 MB

    filter_to_bf16<<<(1024 * KTOT + 255) / 256, 256, 0, stream>>>(tf, Abf);

    dim3 grid(27 /*N tiles*/, 2 /*M tiles*/, 16 /*b = m*4+g*/);
    deform_wmma<<<grid, 256, 0, stream>>>(x, Abf, off, outp);
}
